// FeatureFusionModel_23330262351963
// MI455X (gfx1250) — matrix operand order repaired and checked by reading, not compiled, not run on hardware
//
#include <hip/hip_runtime.h>
#include <hip/hip_bf16.h>

typedef __attribute__((ext_vector_type(16))) __bf16 v16bf;
typedef __attribute__((ext_vector_type(8)))  __bf16 bf16x8;
typedef __attribute__((ext_vector_type(8)))  float   v8f;

#define BATCH   2
#define NVIEWS  6
#define MTOK    1369
#define DIM     384
#define NVOX    50000
#define PF      64
#define HID     256
#define OUTD    16
#define FDIM    448      // PF + DIM
#define LDA     456      // padded row stride of fused-feature tile (bf16)
#define LDH     264      // padded row stride of hidden tile (bf16)
#define TILE_M  64       // voxels per workgroup (4 WMMA M-subtiles)
#define NSUB    4        // TILE_M / 16
#define PGRID   37       // 518 / 14
#define TOTALV  (BATCH * NVOX)

// ---------------------------------------------------------------------------
// Prep kernel 1: mean over views -> bf16 lookup table  [B][MTOK][DIM]
// (mean commutes with the per-voxel gather, so gather traffic drops 6x and the
//  2.1MB bf16 table stays resident in L2 / WGP$)
// ---------------------------------------------------------------------------
__global__ void ff_mean_tokens(const float* __restrict__ patch,
                               __bf16* __restrict__ meanT) {
  int e = blockIdx.x * blockDim.x + threadIdx.x;
  const int per_b = MTOK * DIM;
  if (e >= BATCH * per_b) return;
  int b = e / per_b;
  int r = e - b * per_b;
  const float* p = patch + (size_t)b * NVIEWS * per_b + r;
  float s = 0.f;
#pragma unroll
  for (int v = 0; v < NVIEWS; ++v) s += p[(size_t)v * per_b];
  meanT[e] = (__bf16)(s * (1.0f / 6.0f));
}

// Prep kernel 2: W1 [K=448][N=256] f32 -> W1^T [N][K] bf16 (contiguous-K rows
// so a wave's WMMA B fragment is two contiguous 16B loads per lane)
__global__ void ff_w1t(const float* __restrict__ W1, __bf16* __restrict__ W1T) {
  int e = blockIdx.x * blockDim.x + threadIdx.x;   // e = n*FDIM + k
  if (e >= HID * FDIM) return;
  int n = e / FDIM;
  int k = e - n * FDIM;
  W1T[e] = (__bf16)W1[k * HID + n];
}

// Prep kernel 3: W2 [K=256][N=16] f32 -> W2^T [N][K] bf16
__global__ void ff_w2t(const float* __restrict__ W2, __bf16* __restrict__ W2T) {
  int e = blockIdx.x * blockDim.x + threadIdx.x;   // e = n*HID + k
  if (e >= OUTD * HID) return;
  int n = e / HID;
  int k = e - n * HID;
  W2T[e] = (__bf16)W2[k * OUTD + n];
}

// Build a 16-element WMMA fragment from two contiguous 8-element halves.
__device__ inline v16bf make_frag(bf16x8 lo, bf16x8 hi) {
  v16bf f;
#pragma unroll
  for (int i = 0; i < 8; ++i) { f[i] = lo[i]; f[i + 8] = hi[i]; }
  return f;
}

// ---------------------------------------------------------------------------
// Fused kernel: projection + gather + concat + 448->256 ReLU + 256->16
//   block = 512 threads (16 wave32), 64 voxels per block.
//   Each wave owns a 16-col slice of HIDDEN; a B fragment (W1^T from L2) is
//   reused across 4 M-subtiles. A fragments for all 4 subtiles are loaded
//   into independent live ranges first so the 8 ds_load_b128 issue as one
//   clause and dscnt drains once per k-step (not once per WMMA).
// ---------------------------------------------------------------------------
__global__ __launch_bounds__(512, 1)
void ff_fused_mlp(const float* __restrict__ voxf,
                  const float* __restrict__ voxc,
                  const float* __restrict__ Kmat,
                  const float* __restrict__ Rtm,
                  const __bf16* __restrict__ W1T,
                  const float* __restrict__ b1,
                  const __bf16* __restrict__ W2T,
                  const float* __restrict__ b2,
                  const __bf16* __restrict__ meanT,
                  float* __restrict__ out) {
  __shared__ __bf16 tileA[TILE_M * LDA];   // fused features, bf16 (58.4 KB)
  __shared__ __bf16 tileH[TILE_M * LDH];   // hidden activations, bf16 (33.8 KB)
  __shared__ int s_idx[TILE_M];
  __shared__ int s_b[TILE_M];

  const int tid  = threadIdx.x;
  const int base = blockIdx.x * TILE_M;    // first global voxel of this tile

  // ---- Phase A1: per-voxel projection -> token index (one thread per row)
  if (tid < TILE_M) {
    int gv = base + tid;
    if (gv >= TOTALV) gv = TOTALV - 1;
    int b = gv / NVOX;
    const float* c = voxc + (size_t)gv * 3;
    float x = c[0], y = c[1], z = c[2];
    const float* R = Rtm + b * 12;
    float cx = R[0]*x + R[1]*y + R[2]*z  + R[3];
    float cy = R[4]*x + R[5]*y + R[6]*z  + R[7];
    float cz = R[8]*x + R[9]*y + R[10]*z + R[11];
    const float* Km = Kmat + b * 9;
    float px = Km[0]*cx + Km[1]*cy + Km[2]*cz;
    float py = Km[3]*cx + Km[4]*cy + Km[5]*cz;
    float pz = Km[6]*cx + Km[7]*cy + Km[8]*cz;
    float inv = 1.0f / (pz + 1e-6f);
    float u  = px * inv;
    float vv = py * inv;
    const float su = 518.0f / 1280.0f;
    const float sv = 518.0f / 720.0f;
    int gx = (int)(u  * su * (1.0f / 14.0f));   // trunc toward zero == astype(int32)
    int gy = (int)(vv * sv * (1.0f / 14.0f));
    gx = gx < 0 ? 0 : (gx > PGRID - 1 ? PGRID - 1 : gx);
    gy = gy < 0 ? 0 : (gy > PGRID - 1 ? PGRID - 1 : gy);
    s_idx[tid] = gx * PGRID + gy;
    s_b[tid]   = b;
  }
  __syncthreads();

  // ---- Phase A2: build fused bf16 tile [64][448] (8 threads per row)
  {
    const int m  = tid >> 3;      // row 0..63
    const int t8 = tid & 7;
    int gv = base + m;
    if (gv >= TOTALV) gv = TOTALV - 1;
    // voxel features: 8 f32 per thread -> bf16 (two float4 loads)
    const float4* vfp = (const float4*)(voxf + (size_t)gv * PF) + t8 * 2;
    const float4 v0 = vfp[0];
    const float4 v1 = vfp[1];
    bf16x8 vb;
    vb[0] = (__bf16)v0.x; vb[1] = (__bf16)v0.y;
    vb[2] = (__bf16)v0.z; vb[3] = (__bf16)v0.w;
    vb[4] = (__bf16)v1.x; vb[5] = (__bf16)v1.y;
    vb[6] = (__bf16)v1.z; vb[7] = (__bf16)v1.w;
    *(bf16x8*)&tileA[m * LDA + t8 * 8] = vb;
    // gathered mean tokens: 48 bf16 per thread (6 x 16B copies)
    const bf16x8* src = (const bf16x8*)(meanT +
        ((size_t)s_b[m] * MTOK + s_idx[m]) * DIM) + t8 * 6;
    bf16x8* dst = (bf16x8*)&tileA[m * LDA + PF] + t8 * 6;
#pragma unroll
    for (int c = 0; c < 6; ++c) dst[c] = src[c];
  }
  __syncthreads();

  const int wave  = tid >> 5;     // 0..15
  const int lane  = tid & 31;
  const int lhalf = lane >> 4;    // 0/1: which half-lane group
  const int lm    = lane & 15;

  // ---- Phase B: GEMM1  h = relu(A[64x448] @ W1 + b1), each wave owns 16 cols
  {
    const int n0 = wave * 16;
    v8f acc[NSUB];
#pragma unroll
    for (int s = 0; s < NSUB; ++s)
      acc[s] = (v8f){0.f,0.f,0.f,0.f,0.f,0.f,0.f,0.f};
    // B fragment: lane column n0+lm, k = k0 + lhalf*8 + {0..7, 16..23}
    const __bf16* bb = W1T + (size_t)(n0 + lm) * FDIM + lhalf * 8;
    const int aoff = lhalf * 8;   // A fragment: k = k0 + aoff + {0..7, 16..23}
#pragma unroll
    for (int k0 = 0; k0 < FDIM; k0 += 32) {
      v16bf bfrag = make_frag(*(const bf16x8*)(bb + k0),
                              *(const bf16x8*)(bb + k0 + 16));
      // load all 4 A fragments first (independent live ranges -> one clause)
      v16bf afrag[NSUB];
#pragma unroll
      for (int s = 0; s < NSUB; ++s) {
        const __bf16* ap = &tileA[(s * 16 + lm) * LDA + k0 + aoff];
        afrag[s] = make_frag(*(const bf16x8*)ap,
                             *(const bf16x8*)(ap + 16));
      }
#pragma unroll
      for (int s = 0; s < NSUB; ++s) {
        acc[s] = __builtin_amdgcn_wmma_f32_16x16x32_bf16(
            false, afrag[s], false, bfrag, (short)0, acc[s], false, false);
      }
    }
    const float bias = b1[n0 + lm];
#pragma unroll
    for (int s = 0; s < NSUB; ++s) {
#pragma unroll
      for (int r = 0; r < 8; ++r) {
        const int m0 = r + lhalf * 8;          // C layout: m = r + 8*halflane
        float h = acc[s][r] + bias;
        h = h > 0.f ? h : 0.f;
        tileH[(s * 16 + m0) * LDH + n0 + lm] = (__bf16)h;
      }
    }
  }
  __syncthreads();

  // ---- Phase C: GEMM2  out = h[16x256] @ W2 + b2 (waves 0..3: one subtile each)
  if (wave < NSUB) {
    const int s = wave;
    v8f acc = {0.f,0.f,0.f,0.f,0.f,0.f,0.f,0.f};
    const __bf16* bb = W2T + (size_t)lm * HID + lhalf * 8;
    const int aoff = lhalf * 8;
#pragma unroll
    for (int k0 = 0; k0 < HID; k0 += 32) {
      v16bf bfrag = make_frag(*(const bf16x8*)(bb + k0),
                              *(const bf16x8*)(bb + k0 + 16));
      const __bf16* ap = &tileH[(s * 16 + lm) * LDH + k0 + aoff];
      v16bf afrag = make_frag(*(const bf16x8*)ap,
                              *(const bf16x8*)(ap + 16));
      acc = __builtin_amdgcn_wmma_f32_16x16x32_bf16(
          false, afrag, false, bfrag, (short)0, acc, false, false);
    }
    const float bias = b2[lm];
#pragma unroll
    for (int r = 0; r < 8; ++r) {
      const int m0 = r + lhalf * 8;
      int gv = base + s * 16 + m0;
      if (gv < TOTALV) out[(size_t)gv * OUTD + lm] = acc[r] + bias;
    }
  }
}

// ---------------------------------------------------------------------------
extern "C" void kernel_launch(void* const* d_in, const int* in_sizes, int n_in,
                              void* d_out, int out_size, void* d_ws, size_t ws_size,
                              hipStream_t stream) {
  const float* patch = (const float*)d_in[0];
  const float* voxf  = (const float*)d_in[1];
  const float* voxc  = (const float*)d_in[2];
  const float* Kmat  = (const float*)d_in[3];
  const float* Rtm   = (const float*)d_in[4];
  const float* W1    = (const float*)d_in[5];
  const float* b1    = (const float*)d_in[6];
  const float* W2    = (const float*)d_in[7];
  const float* b2    = (const float*)d_in[8];

  // workspace layout (all 16B aligned): mean LUT | W1^T bf16 | W2^T bf16
  __bf16* meanT = (__bf16*)d_ws;
  __bf16* W1T   = meanT + (size_t)BATCH * MTOK * DIM;  // 1,051,392 elems
  __bf16* W2T   = W1T + (size_t)HID * FDIM;            // 114,688 elems

  {
    int n = BATCH * MTOK * DIM;
    ff_mean_tokens<<<(n + 255) / 256, 256, 0, stream>>>(patch, meanT);
  }
  {
    int n = HID * FDIM;
    ff_w1t<<<(n + 255) / 256, 256, 0, stream>>>(W1, W1T);
  }
  {
    int n = OUTD * HID;
    ff_w2t<<<(n + 255) / 256, 256, 0, stream>>>(W2, W2T);
  }
  ff_fused_mlp<<<(TOTALV + TILE_M - 1) / TILE_M, 512, 0, stream>>>(
      voxf, voxc, Kmat, Rtm, W1T, b1, W2T, b2, meanT, (float*)d_out);
}
